// MSDeformableAttention_20864951124580
// MI455X (gfx1250) — matrix operand order repaired and checked by reading, not compiled, not run on hardware
//
#include <hip/hip_runtime.h>
#include <hip/hip_bf16.h>

// ---------------------------------------------------------------------------
// MS-Deformable-Attention for MI455X (gfx1250, wave32, WMMA bf16->f32)
//
// Pipeline:
//   [1] convert query / input_flatten f32 -> bf16 (row-major M x 256)
//   [2] convert+transpose weights f32 (256 x N) -> bf16 (N x 256)
//   [3] WMMA GEMM value  = input @ W_val + b_val           -> bf16 (B,Len,256)
//   [4] WMMA GEMM offs   = query @ W_off + b_off           -> f32  (B,Lq,256)
//       WMMA GEMM attn   = query @ W_attn + b_attn         -> f32  (B,Lq,128)
//   [5] prep: per (b,q,h) softmax(attn[16]) in-place; offs -> pixel coords
//   [6] sample: wave per (b,q,h), lane = channel, 16-pt bilinear gather
//   [7] WMMA GEMM out    = core @ W_out + b_out            -> f32 d_out
//
// GEMM v2: block stages the 64x256 bf16 weight panel (32KB) into LDS via the
// gfx1250 async-to-LDS path (ASYNCcnt) once; each wave then computes a 64x64
// macro-tile (4 M-tiles x 4 N-tiles, 16 f32 accumulators) with A streamed
// from global (b128 clauses + global_prefetch_b8) and B from ds_load_b128.
// ---------------------------------------------------------------------------

typedef __attribute__((ext_vector_type(16))) __bf16 v16bf;
typedef __attribute__((ext_vector_type(8)))  __bf16 v8bf;
typedef __attribute__((ext_vector_type(8)))  float  v8f;
typedef int v4i __attribute__((vector_size(16)));     // matches builtin param

#define LQ      21760
#define NB      2
#define MQ      (NB * LQ)       // 43520 rows for every GEMM
#define KD      256             // d_model (GEMM K)
#define NHEADS  8
#define HDIM    32
#define MBLOCKS 85              // 43520 / 512 rows per block

#if defined(__has_builtin)
#if __has_builtin(__builtin_amdgcn_global_load_async_to_lds_b128) && \
    __has_builtin(__builtin_amdgcn_s_wait_asynccnt)
#define HAVE_ASYNC_LDS 1
#endif
#endif

typedef __attribute__((address_space(1))) v4i* gv4i_p;   // global v4i*
typedef __attribute__((address_space(3))) v4i* lv4i_p;   // LDS v4i*

// ---------------------------------------------------------------------------
// [1] f32 -> bf16 flat conversion
// ---------------------------------------------------------------------------
__global__ __launch_bounds__(256) void k_f32_to_bf16(const float* __restrict__ in,
                                                     __bf16* __restrict__ out, int n) {
    int i = blockIdx.x * 256 + threadIdx.x;
    if (i < n) out[i] = (__bf16)in[i];
}

// ---------------------------------------------------------------------------
// [2] W (K=256 rows, N cols, row-major) -> WT bf16 (N rows, 256 cols)
// ---------------------------------------------------------------------------
__global__ __launch_bounds__(256) void k_transpose_w(const float* __restrict__ W,
                                                     __bf16* __restrict__ WT, int N) {
    int i = blockIdx.x * 256 + threadIdx.x;
    if (i < KD * N) {
        int k = i / N, n = i % N;
        WT[n * KD + k] = (__bf16)W[i];
    }
}

// ---------------------------------------------------------------------------
// WMMA GEMM:  C[M x N] = A[M x 256] * B[256 x N] + bias
//   A: bf16 row-major.  BT: bf16 transposed weight (N x 256) row-major.
//   Block = 8 waves; block tile = 512(M) x 64(N); wave tile = 64 x 64.
//   M % 512 == 0, N % 64 == 0 (exact here) -> no bounds checks, EXEC all-ones.
//
// Fragment layouts (wave32, 16-bit); A/B K order is contiguous here, not the ISA 7.12.2 table order:
//   A 16x32: lane L: m=L%16, g=L/16; v16bf elems e = K 16g+e
//            (32B contiguous, same order as B)
//   B 32x16: lane L: n=L%16, g=L/16; v16bf elems e = K 16g+e (32B contiguous)
//   C/D f32: VGPR r: row = r + 8g, col = lane%16
// ---------------------------------------------------------------------------
template<int N, bool OUT_BF16>
__global__ __launch_bounds__(256) void k_gemm_bf16(const __bf16* __restrict__ A,
                                                   const __bf16* __restrict__ BT,
                                                   const float*  __restrict__ bias,
                                                   void* __restrict__ Cout) {
    __shared__ __align__(16) __bf16 sB[64 * KD];   // 32KB weight panel

    const int tid   = threadIdx.x;
    const int wave  = tid >> 5;
    const int lane  = tid & 31;
    const int g     = lane >> 4;        // lane group 0/1
    const int l16   = lane & 15;
    const int ntile = N / 64;
    const int nbase = (blockIdx.x % ntile) * 64;
    const int mbw   = (blockIdx.x / ntile) * 512 + wave * 64;   // wave macro-tile

    // ---- stage B panel (rows nbase..nbase+63 of BT) into LDS -------------
    {
        const char* gsrc = (const char*)BT + (size_t)nbase * (KD * 2);
        char*       ldst = (char*)sB;
        // 64 rows * 512B = 32KB = 2048 x 16B chunks; 256 threads -> 8 iters
#pragma unroll
        for (int it = 0; it < 8; ++it) {
            const int c = it * 256 + tid;
#ifdef HAVE_ASYNC_LDS
            __builtin_amdgcn_global_load_async_to_lds_b128(
                (gv4i_p)(gsrc + (size_t)c * 16),
                (lv4i_p)(ldst + (size_t)c * 16),
                0, 0);
#else
            *(uint4*)(ldst + (size_t)c * 16) = *(const uint4*)(gsrc + (size_t)c * 16);
#endif
        }
#ifdef HAVE_ASYNC_LDS
        __builtin_amdgcn_s_wait_asynccnt(0);
#endif
        __syncthreads();
    }

    // ---- compute: 4 M-tiles x 4 N-tiles, K = 256 in 8 steps --------------
    const __bf16* aptr = A + (size_t)(mbw + l16) * KD + g * 16; // + i*4096 + kb
    const __bf16* bls  = sB + (size_t)l16 * KD + g * 16;        // + t*16*KD + kb

    v8f acc[4][4];
#pragma unroll
    for (int i = 0; i < 4; ++i)
#pragma unroll
        for (int t = 0; t < 4; ++t) acc[i][t] = (v8f){};

#pragma unroll
    for (int kb = 0; kb < KD; kb += 32) {
        if (kb + 32 < KD) {            // prefetch next K-slab of A stream
            __builtin_prefetch(aptr + kb + 32, 0, 1);
            __builtin_prefetch(aptr + 3 * 4096 + kb + 32, 0, 1);
        }
        v16bf a[4];
#pragma unroll
        for (int i = 0; i < 4; ++i) {
            a[i] = *(const v16bf*)(aptr + i * 4096 + kb);          // K kb+16g..
        }
#pragma unroll
        for (int t = 0; t < 4; ++t) {
            v16bf b = *(const v16bf*)(bls + (size_t)t * 16 * KD + kb);
#pragma unroll
            for (int i = 0; i < 4; ++i)
                acc[i][t] = __builtin_amdgcn_wmma_f32_16x16x32_bf16(
                    false, a[i], false, b, (short)0, acc[i][t], false, false);
        }
    }

    // ---- epilogue: bias + store ------------------------------------------
#pragma unroll
    for (int i = 0; i < 4; ++i) {
#pragma unroll
        for (int t = 0; t < 4; ++t) {
            const int col = nbase + t * 16 + l16;
            const float bv = bias[col];
#pragma unroll
            for (int r = 0; r < 8; ++r) {
                const size_t idx = (size_t)(mbw + i * 16 + r + g * 8) * N + col;
                const float v = acc[i][t][r] + bv;
                if (OUT_BF16) ((__bf16*)Cout)[idx] = (__bf16)v;
                else          ((float*)Cout)[idx]  = v;
            }
        }
    }
}

// ---------------------------------------------------------------------------
// [5] prep: one thread per (b*Lq, head).
//   attn[b,q, h*16 + l*4 + p] : softmax over 16, in-place
//   offs[b,q, (h*16 + l*4 + p)*2 + {0,1}] : -> absolute pixel coords, in-place
//     loc = ref[l] + off / (W,H);  ix = loc_x * W - 0.5 ; iy = loc_y * H - 0.5
// ---------------------------------------------------------------------------
__global__ __launch_bounds__(256) void k_prep(const float* __restrict__ refp,
                                              float* __restrict__ offs,
                                              float* __restrict__ attn) {
    int t = blockIdx.x * 256 + threadIdx.x;
    if (t >= MQ * NHEADS) return;
    int bq = t >> 3, h = t & 7;

    float* a = attn + (size_t)bq * 128 + h * 16;
    float mx = -1e30f;
#pragma unroll
    for (int i = 0; i < 16; ++i) mx = fmaxf(mx, a[i]);
    float e[16], s = 0.f;
#pragma unroll
    for (int i = 0; i < 16; ++i) { e[i] = __expf(a[i] - mx); s += e[i]; }
    float inv = 1.f / s;
#pragma unroll
    for (int i = 0; i < 16; ++i) a[i] = e[i] * inv;

    const float dimv[4] = {128.f, 64.f, 32.f, 16.f};   // square levels (W == H)
    const float* rp = refp + (size_t)bq * 8;           // (4 levels, 2)
    float* o = offs + (size_t)bq * 256 + h * 32;       // 16 points * 2
#pragma unroll
    for (int l = 0; l < 4; ++l) {
        const float d = dimv[l], invd = 1.f / d;
        const float rx = rp[l * 2 + 0], ry = rp[l * 2 + 1];
#pragma unroll
        for (int p = 0; p < 4; ++p) {
            float* op = o + (l * 4 + p) * 2;
            const float lx = rx + op[0] * invd;
            const float ly = ry + op[1] * invd;
            op[0] = lx * d - 0.5f;
            op[1] = ly * d - 0.5f;
        }
    }
}

// ---------------------------------------------------------------------------
// [6] sampling core: one wave per (b,q,h); lane = channel d in [0,32).
//   value: bf16 (B, Len=21760, 256), channel = h*32 + d
//   Each corner read is 32 consecutive bf16 across the wave (coalesced 64B).
// ---------------------------------------------------------------------------
__global__ __launch_bounds__(256) void k_sample(const __bf16* __restrict__ value,
                                                const float* __restrict__ loc,
                                                const float* __restrict__ attn,
                                                __bf16* __restrict__ core) {
    const int wave = threadIdx.x >> 5;
    const int lane = threadIdx.x & 31;
    const int unit = blockIdx.x * 8 + wave;      // bq * 8 + h
    if (unit >= MQ * NHEADS) return;
    const int bq = unit >> 3, h = unit & 7;
    const int b  = bq / LQ;

    const float* lo = loc  + (size_t)bq * 256 + h * 32;
    const float* aw = attn + (size_t)bq * 128 + h * 16;
    const __bf16* vb = value + ((size_t)b * LQ) * KD + h * HDIM + lane;

    const int loff[4] = {0, 16384, 20480, 21504};
    const int dimi[4] = {128, 64, 32, 16};

    float acc = 0.f;
#pragma unroll
    for (int l = 0; l < 4; ++l) {
        const int W = dimi[l], base = loff[l];
#pragma unroll
        for (int p = 0; p < 4; ++p) {
            const float ix = lo[(l * 4 + p) * 2 + 0];
            const float iy = lo[(l * 4 + p) * 2 + 1];
            const float w  = aw[l * 4 + p];
            const int x0 = (int)floorf(ix), y0 = (int)floorf(iy);
            const int x1 = x0 + 1, y1 = y0 + 1;
            const float fx = ix - (float)x0, fy = iy - (float)y0;
            const float w00 = (1.f - fx) * (1.f - fy) * w;
            const float w01 = fx * (1.f - fy) * w;
            const float w10 = (1.f - fx) * fy * w;
            const float w11 = fx * fy * w;
            if (x0 >= 0 && x0 < W && y0 >= 0 && y0 < W)
                acc += w00 * (float)vb[(size_t)(base + y0 * W + x0) * KD];
            if (x1 >= 0 && x1 < W && y0 >= 0 && y0 < W)
                acc += w01 * (float)vb[(size_t)(base + y0 * W + x1) * KD];
            if (x0 >= 0 && x0 < W && y1 >= 0 && y1 < W)
                acc += w10 * (float)vb[(size_t)(base + y1 * W + x0) * KD];
            if (x1 >= 0 && x1 < W && y1 >= 0 && y1 < W)
                acc += w11 * (float)vb[(size_t)(base + y1 * W + x1) * KD];
        }
    }
    core[(size_t)bq * KD + h * HDIM + lane] = (__bf16)acc;
}

// ---------------------------------------------------------------------------
extern "C" void kernel_launch(void* const* d_in, const int* in_sizes, int n_in,
                              void* d_out, int out_size, void* d_ws, size_t ws_size,
                              hipStream_t stream) {
    const float* query  = (const float*)d_in[0];
    const float* refp   = (const float*)d_in[1];
    const float* inputf = (const float*)d_in[2];
    const float* W_off  = (const float*)d_in[3];
    const float* b_off  = (const float*)d_in[4];
    const float* W_attn = (const float*)d_in[5];
    const float* b_attn = (const float*)d_in[6];
    const float* W_val  = (const float*)d_in[7];
    const float* b_val  = (const float*)d_in[8];
    const float* W_out  = (const float*)d_in[9];
    const float* b_out  = (const float*)d_in[10];

    // workspace carve-out (256B aligned)
    char* ws = (char*)d_ws;
    size_t o = 0;
    auto alloc = [&](size_t bytes) -> char* {
        o = (o + 255) & ~(size_t)255;
        char* r = ws + o;
        o += bytes;
        return r;
    };
    const size_t SZ_M256_BF = (size_t)MQ * KD * 2;
    __bf16* qB     = (__bf16*)alloc(SZ_M256_BF);           // query bf16
    __bf16* vB     = (__bf16*)alloc(SZ_M256_BF);           // input_flatten bf16
    __bf16* valB   = (__bf16*)alloc(SZ_M256_BF);           // value bf16
    __bf16* coreB  = (__bf16*)alloc(SZ_M256_BF);           // sampling output bf16
    __bf16* wtVal  = (__bf16*)alloc((size_t)KD * 256 * 2); // transposed weights
    __bf16* wtOff  = (__bf16*)alloc((size_t)KD * 256 * 2);
    __bf16* wtAttn = (__bf16*)alloc((size_t)KD * 128 * 2);
    __bf16* wtOut  = (__bf16*)alloc((size_t)KD * 256 * 2);
    float*  offsF  = (float*)alloc((size_t)MQ * 256 * 4);  // offsets -> pixel coords
    float*  attnF  = (float*)alloc((size_t)MQ * 128 * 4);  // logits -> softmax
    (void)ws_size;

    const int nElem = MQ * KD;                             // 11,141,120
    // [1] activations -> bf16
    k_f32_to_bf16<<<(nElem + 255) / 256, 256, 0, stream>>>(query,  qB, nElem);
    k_f32_to_bf16<<<(nElem + 255) / 256, 256, 0, stream>>>(inputf, vB, nElem);
    // [2] weights -> bf16 transposed
    k_transpose_w<<<(KD * 256 + 255) / 256, 256, 0, stream>>>(W_val,  wtVal,  256);
    k_transpose_w<<<(KD * 256 + 255) / 256, 256, 0, stream>>>(W_off,  wtOff,  256);
    k_transpose_w<<<(KD * 128 + 255) / 256, 256, 0, stream>>>(W_attn, wtAttn, 128);
    k_transpose_w<<<(KD * 256 + 255) / 256, 256, 0, stream>>>(W_out,  wtOut,  256);

    // [3] value projection (bf16 out for the gather path)
    k_gemm_bf16<256, true ><<<MBLOCKS * (256 / 64), 256, 0, stream>>>(vB, wtVal, b_val, valB);
    // [4] offset + attention projections (f32 out)
    k_gemm_bf16<256, false><<<MBLOCKS * (256 / 64), 256, 0, stream>>>(qB, wtOff, b_off, offsF);
    k_gemm_bf16<128, false><<<MBLOCKS * (128 / 64), 256, 0, stream>>>(qB, wtAttn, b_attn, attnF);
    // [5] softmax + pixel-coord prep
    k_prep<<<(MQ * NHEADS + 255) / 256, 256, 0, stream>>>(refp, offsF, attnF);
    // [6] deformable bilinear sampling
    k_sample<<<(MQ * NHEADS + 7) / 8, 256, 0, stream>>>(valB, offsF, attnF, coreB);
    // [7] output projection -> d_out (f32)
    k_gemm_bf16<256, false><<<MBLOCKS * (256 / 64), 256, 0, stream>>>(coreB, wtOut, b_out, (float*)d_out);
}
